// UserEncode_59717225283584
// MI455X (gfx1250) — matrix operand order repaired and checked by reading, not compiled, not run on hardware
//
#include <hip/hip_runtime.h>
#include <hip/hip_bf16.h>
#include <math.h>

typedef __attribute__((ext_vector_type(16))) _Float16 v16h;
typedef __attribute__((ext_vector_type(8)))  float    v8f;

#define N_NODES_C  16384
#define HIST_LEN_C 50
#define DIM_C      64

// LDS row strides in halfs (padded +8 halfs = 16B so row-strided accesses
// from 16 lanes land on distinct bank groups)
#define XS 136   // 128 + 8
#define HS 72    // 64 + 8

union H16 { v16h v; uint4 q[2]; };
union H8  { _Float16 h[8]; uint4 q; };

// f16 weight layout in d_ws (in halfs):
//   [    0,  8192)  W1  [64][128]
//   [ 8192, 12288)  W2  [64][64]
//   [12288, 20480)  A1  [64][128]
//   [20480, 24576)  A2  [64][64]
__global__ void graphrec_convert_weights(const float* __restrict__ w1,
                                         const float* __restrict__ w2,
                                         const float* __restrict__ a1,
                                         const float* __restrict__ a2,
                                         _Float16* __restrict__ wf) {
  int i = blockIdx.x * blockDim.x + threadIdx.x;
  if (i < 8192)       wf[i] = (_Float16)w1[i];
  else if (i < 12288) wf[i] = (_Float16)w2[i - 8192];
  else if (i < 20480) wf[i] = (_Float16)a1[i - 12288];
  else if (i < 24576) wf[i] = (_Float16)a2[i - 20480];
}

// --- WMMA fragment helpers (CDNA5 16x16x32 f16 layouts, wave32) ------------

// A (16x32, MxK): lanes 0-15 -> M=lane, halfs 0..15 = K0..15
//                 lanes 16-31 -> M=lane-16, halfs 0..15 = K16..31
__device__ __forceinline__ v16h ldsA(const _Float16* X, int stride, int m,
                                     int kk, int lane) {
  int hi = (lane >> 4) & 1;
  const _Float16* p = X + m * stride + kk * 32 + hi * 16;
  H16 r;
  r.q[0] = *(const uint4*)p;         // 8 halfs
  r.q[1] = *(const uint4*)(p + 8);   // 8 halfs, +8 K
  return r.v;
}

// B (32x16, KxN): col N = lane&15; lanes 0-15 hold K=0..15, lanes 16-31 K=16..31
// Weight matrix Wf is [64][kdim] row-major (out, in): B[k][j] = Wf[j][k]
__device__ __forceinline__ v16h gB(const _Float16* __restrict__ Wf, int kdim,
                                   int jj, int kk, int lane) {
  int j  = jj * 16 + (lane & 15);
  int hi = (lane >> 4) & 1;
  const _Float16* p = Wf + j * kdim + kk * 32 + hi * 16;
  H16 r;
  r.q[0] = *(const uint4*)p;
  r.q[1] = *(const uint4*)(p + 8);
  return r.v;
}

// att3 "matrix": only column 0 carries w3, rest zero
__device__ __forceinline__ v16h w3frag(const float* __restrict__ w3, int kk,
                                       int lane) {
  v16h b = {};
  if ((lane & 15) == 0) {
    int kb = kk * 32 + ((lane & 16) ? 16 : 0);
#pragma unroll
    for (int h = 0; h < 16; ++h) b[h] = (_Float16)w3[kb + h];
  }
  return b;
}

// D/C (16x16 f32): VGPR r -> lanes 0-15: (M=r, N=lane); lanes 16-31: (M=8+r, N=lane-16)
__device__ __forceinline__ void storeD_relu(v8f c, _Float16* Y, int stride,
                                            int rowbase, int jj, int lane,
                                            const float* __restrict__ bias) {
  int col = jj * 16 + (lane & 15);
  float bv = bias[col];
  int rb = rowbase + ((lane & 16) ? 8 : 0);
#pragma unroll
  for (int r = 0; r < 8; ++r) {
    float v = c[r] + bv;
    Y[(rb + r) * stride + col] = (_Float16)(v > 0.f ? v : 0.f);
  }
}

// One dense layer for this wave's 16 rows: Y = relu(X @ Wf^T + bias).
// Reads its A fragments into registers up-front, so in-place X==Y is safe.
template <int KDIM>
__device__ __forceinline__ void layerMLP(const _Float16* X, int xs,
                                         _Float16* Y, int ys,
                                         const _Float16* __restrict__ Wf,
                                         const float* __restrict__ bias,
                                         int rowbase, int lane) {
  constexpr int NK = KDIM / 32;
  v16h a[NK];
  int m = rowbase + (lane & 15);
#pragma unroll
  for (int kk = 0; kk < NK; ++kk) a[kk] = ldsA(X, xs, m, kk, lane);
#pragma unroll
  for (int jj = 0; jj < 4; ++jj) {
    v8f c = {};
#pragma unroll
    for (int kk = 0; kk < NK; ++kk) {
      v16h b = gB(Wf, KDIM, jj, kk, lane);
      c = __builtin_amdgcn_wmma_f32_16x16x32_f16(false, a[kk], false, b,
                                                 (short)0, c, false, false);
    }
    storeD_relu(c, Y, ys, rowbase, jj, lane, bias);
  }
}

__launch_bounds__(128)
__global__ void graphrec_user_encode(
    const int* __restrict__ nodes,
    const int* __restrict__ hist_items,
    const int* __restrict__ hist_ratings,
    const int* __restrict__ hist_len,
    const float* __restrict__ u2e,
    const float* __restrict__ i2e,
    const float* __restrict__ r2e,
    const _Float16* __restrict__ wf,
    const float* __restrict__ w1_b,
    const float* __restrict__ w2_b,
    const float* __restrict__ a1_b,
    const float* __restrict__ a2_b,
    const float* __restrict__ w3,
    const float* __restrict__ b3,
    float* __restrict__ out) {
  __shared__ __align__(16) _Float16 X1[64 * XS];  // layer-1 input / [o ; u]
  __shared__ __align__(16) _Float16 Hb[64 * HS];  // hidden buffer
  __shared__ __align__(16) _Float16 uf[64];
  __shared__ float scores[64];
  __shared__ float attw[64];

  const int tid = threadIdx.x;
  const int lane = tid & 31;
  const int wave = tid >> 5;
  const int rowbase = wave * 16;
  const int nblk = blockIdx.x;
  const int node = nodes[nblk];

  // Phase 0: gather [p_embed ; r_embed] -> f16 X1; u_rep -> f16 uf
  {
    int row = tid >> 1, half = tid & 1;
    _Float16* dst = &X1[row * XS + half * 64];
    if (row < HIST_LEN_C) {
      int idx = half ? hist_ratings[nblk * HIST_LEN_C + row]
                     : hist_items[nblk * HIST_LEN_C + row];
      const float* src = (half ? r2e : i2e) + (size_t)idx * DIM_C;
#pragma unroll
      for (int c = 0; c < 64; c += 8) {
        float4 f0 = *(const float4*)(src + c);
        float4 f1 = *(const float4*)(src + c + 4);
        H8 h;
        h.h[0] = (_Float16)f0.x; h.h[1] = (_Float16)f0.y;
        h.h[2] = (_Float16)f0.z; h.h[3] = (_Float16)f0.w;
        h.h[4] = (_Float16)f1.x; h.h[5] = (_Float16)f1.y;
        h.h[6] = (_Float16)f1.z; h.h[7] = (_Float16)f1.w;
        *(uint4*)(dst + c) = h.q;
      }
    } else {  // pad rows 50..63 with zeros
      uint4 z{0, 0, 0, 0};
#pragma unroll
      for (int c = 0; c < 64; c += 8) *(uint4*)(dst + c) = z;
    }
    if (tid < 16) {
      float4 f = *(const float4*)(u2e + (size_t)node * DIM_C + tid * 4);
      H8 h;
      h.h[0] = (_Float16)f.x; h.h[1] = (_Float16)f.y;
      h.h[2] = (_Float16)f.z; h.h[3] = (_Float16)f.w;
      *(uint2*)&uf[tid * 4] = *(uint2*)&h.h[0];
    }
  }
  __syncthreads();

  // MLP1: X1[:,0:128] -> Hb     (16 WMMA / wave)
  layerMLP<128>(X1, XS, Hb, HS, wf + 0, w1_b, rowbase, lane);
  __syncthreads();  // all waves done reading X1 before u-fill below

  // broadcast u into X1[:,64:128]; MLP2: Hb -> X1[:,0:64] = o  (8 WMMA / wave)
  {
    int row = tid >> 1, off = (tid & 1) * 32;
    const uint4* s = (const uint4*)&uf[off];
    uint4* d = (uint4*)&X1[row * XS + 64 + off];
    d[0] = s[0]; d[1] = s[1]; d[2] = s[2]; d[3] = s[3];
  }
  layerMLP<64>(Hb, HS, X1, XS, wf + 8192, w2_b, rowbase, lane);
  __syncthreads();

  // Att1: X1 = [o ; u] -> Hb    (16 WMMA / wave)
  layerMLP<128>(X1, XS, Hb, HS, wf + 12288, a1_b, rowbase, lane);
  // Att2: Hb -> Hb in place (safe: wave reads own rows to regs first)  (8 WMMA)
  layerMLP<64>(Hb, HS, Hb, HS, wf + 20480, a2_b, rowbase, lane);

  // Att3 via WMMA: scores land in column 0 -> lanes 0 and 16  (2 WMMA)
  {
    int m = rowbase + (lane & 15);
    v16h a0 = ldsA(Hb, HS, m, 0, lane);
    v16h a1 = ldsA(Hb, HS, m, 1, lane);
    v16h b0 = w3frag(w3, 0, lane);
    v16h b1 = w3frag(w3, 1, lane);
    v8f c = {};
    c = __builtin_amdgcn_wmma_f32_16x16x32_f16(false, a0, false, b0, (short)0,
                                               c, false, false);
    c = __builtin_amdgcn_wmma_f32_16x16x32_f16(false, a1, false, b1, (short)0,
                                               c, false, false);
    if ((lane & 15) == 0) {
      int rb = rowbase + ((lane & 16) ? 8 : 0);
      float bb = b3[0];
#pragma unroll
      for (int r = 0; r < 8; ++r) scores[rb + r] = c[r] + bb;
    }
  }
  __syncthreads();

  // Masked softmax over the ragged history (len in [1,50])
  if (tid == 0) {
    int len = hist_len[nblk];
    float mx = -1e30f;
    for (int l = 0; l < len; ++l) mx = fmaxf(mx, scores[l]);
    float s = 0.f;
    for (int l = 0; l < len; ++l) {
      float e = __expf(scores[l] - mx);
      attw[l] = e;
      s += e;
    }
    float inv = 1.f / s;
    for (int l = 0; l < HIST_LEN_C; ++l)
      attw[l] = (l < len) ? attw[l] * inv : 0.f;
  }
  __syncthreads();

  // out[n][j] = sum_l attw[l] * o[l][j]   (o is f16 in X1[:,0:64])
  if (tid < 64) {
    float acc = 0.f;
#pragma unroll 5
    for (int l = 0; l < HIST_LEN_C; ++l)
      acc += attw[l] * (float)X1[l * XS + tid];
    out[(size_t)nblk * DIM_C + tid] = acc;
  }
}

extern "C" void kernel_launch(void* const* d_in, const int* in_sizes, int n_in,
                              void* d_out, int out_size, void* d_ws,
                              size_t ws_size, hipStream_t stream) {
  (void)in_sizes; (void)n_in; (void)out_size; (void)ws_size;
  const int*   nodes  = (const int*)d_in[0];
  const int*   hitems = (const int*)d_in[1];
  const int*   hrat   = (const int*)d_in[2];
  const int*   hlen   = (const int*)d_in[3];
  const float* u2e    = (const float*)d_in[4];
  const float* i2e    = (const float*)d_in[5];
  const float* r2e    = (const float*)d_in[6];
  const float* w1w    = (const float*)d_in[7];
  const float* w1b    = (const float*)d_in[8];
  const float* w2w    = (const float*)d_in[9];
  const float* w2b    = (const float*)d_in[10];
  const float* a1w    = (const float*)d_in[11];
  const float* a1b    = (const float*)d_in[12];
  const float* a2w    = (const float*)d_in[13];
  const float* a2b    = (const float*)d_in[14];
  const float* w3     = (const float*)d_in[15];
  const float* b3     = (const float*)d_in[16];
  _Float16* wf = (_Float16*)d_ws;
  float* out = (float*)d_out;

  graphrec_convert_weights<<<96, 256, 0, stream>>>(w1w, w2w, a1w, a2w, wf);
  graphrec_user_encode<<<N_NODES_C, 128, 0, stream>>>(
      nodes, hitems, hrat, hlen, u2e, i2e, r2e, wf, w1b, w2b, a1b, a2b, w3, b3,
      out);
}
